// Attention_25185688223835
// MI455X (gfx1250) — matrix operand order repaired and checked by reading, not compiled, not run on hardware
//
#include <hip/hip_runtime.h>
#include <hip/hip_bf16.h>
#include <math.h>

typedef __attribute__((ext_vector_type(16))) __bf16 v16bf;
typedef __attribute__((ext_vector_type(8)))  __bf16 v8bf;
typedef __attribute__((ext_vector_type(8)))  float  v8f;

#define BATCH 32
#define SEQ   2048
#define HDIM  1024

// Fast tanh: native V_TANH_F32 (gfx1250 TRANS op, co-executes with WMMA).
#if defined(__has_builtin)
#if __has_builtin(__builtin_amdgcn_tanhf)
#define HAVE_AMDGCN_TANH 1
#endif
#endif
__device__ __forceinline__ float fast_tanh(float x) {
#ifdef HAVE_AMDGCN_TANH
    return __builtin_amdgcn_tanhf(x);
#else
    float e = __expf(2.0f * x);          // branchless fallback, saturates at +-1
    return 1.0f - 2.0f / (e + 1.0f);
#endif
}

#define WMMA_BF16(a, b, c) __builtin_amdgcn_wmma_f32_16x16x32_bf16( \
        false, (a), false, (b), (short)0, (c), false, false)

// ---------------------------------------------------------------------------
// Kernel 0: zero the logits accumulator (atomics target) each launch.
// ---------------------------------------------------------------------------
__global__ void zero_kernel(float* __restrict__ p, int n) {
    int i = blockIdx.x * blockDim.x + threadIdx.x;
    if (i < n) p[i] = 0.0f;
}

// ---------------------------------------------------------------------------
// Kernel 1: LDS-tiled transpose  WeT[n][k] = (bf16) W_attn[HDIM + k][n]
// ---------------------------------------------------------------------------
__global__ __launch_bounds__(256) void wet_kernel(const float* __restrict__ W,
                                                  __bf16* __restrict__ WeT) {
    __shared__ float tile[32][33];
    const int bx = blockIdx.x;
    const int tk = (bx & 31) * 32;   // k tile base
    const int tn = (bx >> 5) * 32;   // n tile base
    const int tx = threadIdx.x & 31;
    const int ty = threadIdx.x >> 5; // 0..7
    #pragma unroll
    for (int i = 0; i < 32; i += 8)
        tile[ty + i][tx] = W[(size_t)(HDIM + tk + ty + i) * HDIM + tn + tx];
    __syncthreads();
    #pragma unroll
    for (int i = 0; i < 32; i += 8)
        WeT[(size_t)(tn + ty + i) * HDIM + tk + tx] = (__bf16)tile[tx][ty + i];
}

// ---------------------------------------------------------------------------
// Kernel 2: h_proj[b][n] = sum_k hidden[b][k] * Wh[k][n] + b_attn[n]   (f32)
// ---------------------------------------------------------------------------
__global__ __launch_bounds__(256) void hproj_kernel(const float* __restrict__ hidden,
                                                    const float* __restrict__ W,
                                                    const float* __restrict__ b_attn,
                                                    float* __restrict__ h_proj) {
    int idx = blockIdx.x * blockDim.x + threadIdx.x;   // 32*1024
    int b = idx >> 10, n = idx & (HDIM - 1);
    const float* hrow = hidden + (size_t)b * HDIM;
    float acc = b_attn[n];
    for (int k = 0; k < HDIM; ++k)
        acc = fmaf(hrow[k], W[(size_t)k * HDIM + n], acc);
    h_proj[idx] = acc;
}

// ---------------------------------------------------------------------------
// A-tile LDS layout: 32 rows x 1024 bf16 (2048 B/row) = exactly 64 KB, with a
// 16-byte-chunk XOR swizzle (chunk ^= row&15) so ds_load_b128 A-fragments
// (lanes 0-15 = rows 0-15, same K-chunk) are bank-conflict-free.
// ---------------------------------------------------------------------------
__device__ __forceinline__ v16bf load_a_frag(const __bf16* At, int row, int k0, int half) {
    const char* base = (const char*)At + row * 2048;
    const int sw = row & 15;
    const int c0 = (k0 >> 3) + half * 2;   // chunk of elems K k0+half*16 .. +7
    const int c1 = c0 + 1;                 // chunk of elems K k0+half*16+8 .. +7
    v8bf alo = *(const v8bf*)(base + ((c0 ^ sw) << 4));
    v8bf ahi = *(const v8bf*)(base + ((c1 ^ sw) << 4));
    return __builtin_shufflevector(alo, ahi,
                0,1,2,3,4,5,6,7,8,9,10,11,12,13,14,15);
}

// ---------------------------------------------------------------------------
// Kernel 3 (main): fused
//   logits[b,s] = sum_n tanh(enc[b,s,:]@We[:,n] + h_proj[b,n]) * v[n]
// One WG (128 thr = 4 wave32) per 32-row tile. Each wave owns a 256-col
// N-slice and runs a 2x2 (row-tile x n-tile) WMMA grid per k-step: every
// B fragment feeds 2 WMMAs and every A fragment feeds 2 WMMAs.
// ---------------------------------------------------------------------------
__global__ __launch_bounds__(128) void energy_kernel(const float* __restrict__ enc,
                                                     const __bf16* __restrict__ WeT,
                                                     const float* __restrict__ h_proj,
                                                     const float* __restrict__ vvec,
                                                     float* __restrict__ logits) {
    __shared__ __align__(16) __bf16 Atile[32 * HDIM];   // 64 KB, swizzled

    const int row0 = blockIdx.x * 32;   // global row base (b*S + s); within one batch
    const int bidx = row0 / SEQ;
    const int tid  = threadIdx.x;

    // ---- cooperative load: 32 rows x 1024 f32 -> bf16 swizzled LDS tile
    {
        const float4* src = (const float4*)(enc + (size_t)row0 * HDIM);
        for (int i = tid; i < 32 * HDIM / 4; i += 128) {   // 8192 float4
            float4 x = src[i];
            int r = i >> 8;               // 256 float4 per row
            int c = (i & 255) * 4;        // element column (multiple of 4)
            int chunk = c >> 3;           // 16B chunk within row
            int sub   = (c & 7) * 2;      // byte offset inside chunk (0 or 8)
            __bf16* d = (__bf16*)((char*)Atile + r * 2048 +
                                  (((chunk ^ (r & 15)) << 4) + sub));
            d[0] = (__bf16)x.x; d[1] = (__bf16)x.y;
            d[2] = (__bf16)x.z; d[3] = (__bf16)x.w;
        }
    }
    __syncthreads();

    const int wave  = tid >> 5;        // 0..3 -> N-slice of 256 columns
    const int lane  = tid & 31;
    const int l16   = lane & 15;
    const int half  = lane >> 4;       // 0: lanes 0-15, 1: lanes 16-31

    const float* hp_row = h_proj + (size_t)bidx * HDIM;

    float part0[8] = {0.f,0.f,0.f,0.f,0.f,0.f,0.f,0.f};   // rows row0 + 0..15
    float part1[8] = {0.f,0.f,0.f,0.f,0.f,0.f,0.f,0.f};   // rows row0 + 16..31

    for (int nt = 0; nt < 16; nt += 2) {
        const int n0 = wave * 256 + nt * 16;
        // B fragment bases: column n0(+16)+l16, K contiguous; half picks K+0/K+16
        const __bf16* bb0 = WeT + (size_t)(n0 + l16) * HDIM + half * 16;
        const __bf16* bb1 = bb0 + (size_t)16 * HDIM;

        v8f c00 = {}, c01 = {}, c10 = {}, c11 = {};   // c[rowtile][ntile]
        #pragma unroll 2
        for (int kc = 0; kc < 32; ++kc) {
            const int k0 = kc * 32;
            v16bf a0 = load_a_frag(Atile, l16,      k0, half);
            v16bf a1 = load_a_frag(Atile, 16 + l16, k0, half);
            v16bf b0 = *(const v16bf*)(bb0 + k0);
            v16bf b1 = *(const v16bf*)(bb1 + k0);
            c00 = WMMA_BF16(a0, b0, c00);
            c10 = WMMA_BF16(a1, b0, c10);
            c01 = WMMA_BF16(a0, b1, c01);
            c11 = WMMA_BF16(a1, b1, c11);
        }

        // epilogue: C layout — VGPR r, lane: row = r + 8*half, col = l16
        {
            const int n  = n0 + l16;
            const float hp = hp_row[n];
            const float vn = vvec[n];
            #pragma unroll
            for (int r = 0; r < 8; ++r) {
                part0[r] += fast_tanh(c00[r] + hp) * vn;
                part1[r] += fast_tanh(c10[r] + hp) * vn;
            }
        }
        {
            const int n  = n0 + 16 + l16;
            const float hp = hp_row[n];
            const float vn = vvec[n];
            #pragma unroll
            for (int r = 0; r < 8; ++r) {
                part0[r] += fast_tanh(c01[r] + hp) * vn;
                part1[r] += fast_tanh(c11[r] + hp) * vn;
            }
        }
    }

    // reduce over the 16 columns held by each 16-lane half (xor<16 stays in-half)
    #pragma unroll
    for (int r = 0; r < 8; ++r) {
        float p0 = part0[r];
        float p1 = part1[r];
        p0 += __shfl_xor(p0, 1);  p1 += __shfl_xor(p1, 1);
        p0 += __shfl_xor(p0, 2);  p1 += __shfl_xor(p1, 2);
        p0 += __shfl_xor(p0, 4);  p1 += __shfl_xor(p1, 4);
        p0 += __shfl_xor(p0, 8);  p1 += __shfl_xor(p1, 8);
        if (l16 == 0) {
            const int m = r + half * 8;                       // local row 0..15
            atomicAdd(&logits[row0 + m],      p0);            // rows 0..15
            atomicAdd(&logits[row0 + 16 + m], p1);            // rows 16..31
        }
    }
}

// ---------------------------------------------------------------------------
// Kernel 4: softmax over S per batch. One block (256 thr) per batch row.
// ---------------------------------------------------------------------------
__global__ __launch_bounds__(256) void softmax_kernel(const float* __restrict__ logits,
                                                      float* __restrict__ out) {
    __shared__ float red[256];
    const int b = blockIdx.x, tid = threadIdx.x;
    const float* row = logits + (size_t)b * SEQ;

    float m = -3.402823466e38f;
    for (int i = tid; i < SEQ; i += 256) m = fmaxf(m, row[i]);
    red[tid] = m; __syncthreads();
    for (int s = 128; s > 0; s >>= 1) {
        if (tid < s) red[tid] = fmaxf(red[tid], red[tid + s]);
        __syncthreads();
    }
    m = red[0]; __syncthreads();

    float sum = 0.f;
    for (int i = tid; i < SEQ; i += 256) sum += __expf(row[i] - m);
    red[tid] = sum; __syncthreads();
    for (int s = 128; s > 0; s >>= 1) {
        if (tid < s) red[tid] += red[tid + s];
        __syncthreads();
    }
    const float inv = 1.0f / red[0];

    for (int i = tid; i < SEQ; i += 256)
        out[(size_t)b * SEQ + i] = __expf(row[i] - m) * inv;
}

// ---------------------------------------------------------------------------
extern "C" void kernel_launch(void* const* d_in, const int* in_sizes, int n_in,
                              void* d_out, int out_size, void* d_ws, size_t ws_size,
                              hipStream_t stream) {
    const float* hidden = (const float*)d_in[0];   // (B, H)
    const float* enc    = (const float*)d_in[1];   // (B, S, H)
    const float* W_attn = (const float*)d_in[2];   // (2H, H)
    const float* b_attn = (const float*)d_in[3];   // (H,)
    const float* vvec   = (const float*)d_in[4];   // (H,)
    float* out = (float*)d_out;                    // (B, S)

    // workspace layout
    char* ws = (char*)d_ws;
    __bf16* WeT   = (__bf16*)ws;                                   // 2 MB
    float* h_proj = (float*)(ws + (size_t)HDIM * HDIM * 2);        // 128 KB
    float* logits = (float*)(ws + (size_t)HDIM * HDIM * 2
                                + (size_t)BATCH * HDIM * 4);       // 256 KB

    const int nlogits = BATCH * SEQ;

    zero_kernel<<<(nlogits + 255) / 256, 256, 0, stream>>>(logits, nlogits);
    wet_kernel<<<(HDIM / 32) * (HDIM / 32), 256, 0, stream>>>(W_attn, WeT);
    hproj_kernel<<<(BATCH * HDIM) / 256, 256, 0, stream>>>(hidden, W_attn, b_attn, h_proj);
    energy_kernel<<<(BATCH * SEQ) / 32, 128, 0, stream>>>(enc, WeT, h_proj, vvec, logits);
    softmax_kernel<<<BATCH, 256, 0, stream>>>(logits, out);
}
